// Encoder_28630251995589
// MI455X (gfx1250) — matrix operand order repaired and checked by reading, not compiled, not run on hardware
//
#include <hip/hip_runtime.h>

#define D_      512
#define HALF_   256
#define DFF_    2048
#define NEXP_   16
#define TOPK_   4
#define NVIEWS_ 3
#define B_      64
#define L_      8
#define T_      512   // B*L tokens
#define HEADS_  4
#define DH_     128   // D/HEADS
#define NLAYERS_ 3
#define RES_    5000

typedef __attribute__((ext_vector_type(16))) __bf16 v16bf;
typedef __attribute__((ext_vector_type(8)))  float  v8f;

union BF16x16 { unsigned int u[8]; v16bf v; };

__device__ __forceinline__ unsigned short f2bf(float f) {
  unsigned int u = __float_as_uint(f);
  unsigned int r = (u + 0x7FFFu + ((u >> 16) & 1u)) >> 16;   // RNE
  return (unsigned short)r;
}

// ---------------------------------------------------------------- converts
__global__ __launch_bounds__(256) void k_f32_to_bf16(
    const float* __restrict__ s, unsigned short* __restrict__ d, int n) {
  for (int i = blockIdx.x * blockDim.x + threadIdx.x; i < n;
       i += gridDim.x * blockDim.x)
    d[i] = f2bf(s[i]);
}

__global__ __launch_bounds__(64) void k_zero_i32(int* p, int n) {
  int i = threadIdx.x;
  if (i < n) p[i] = 0;
}

// -------------------------------------------------- embedding + projection
// out[t][d] = (sum_m emb[Z[t]][m] * W[d][m] + b[d]) * 2^emb_scale
__global__ __launch_bounds__(256) void k_embed(
    const int* __restrict__ Z, const float* __restrict__ emb, int M,
    const float* __restrict__ W, const float* __restrict__ bias,
    const float* __restrict__ scale_p,
    float* __restrict__ vf, unsigned short* __restrict__ vb) {
  int t = blockIdx.x;
  int z = Z[t];
  __shared__ float se[256];
  for (int i = threadIdx.x; i < M; i += 256) se[i] = emb[z * M + i];
  __syncthreads();
  float s = exp2f(scale_p[0]);
  for (int d = threadIdx.x; d < D_; d += 256) {
    float acc = bias[d];
    const float* wr = W + (size_t)d * M;
    for (int m = 0; m < M; ++m) acc += se[m] * wr[m];
    float val = acc * s;
    vf[(size_t)t * D_ + d] = val;
    vb[(size_t)t * D_ + d] = f2bf(val);
  }
}

// ------------------------------------------- gating: top-4 + expert lists
// logits_e = 2 v.k_e - ||k_e||^2 + r_e.v   (-||v||^2 dropped: shift-invariant)
__global__ __launch_bounds__(32) void k_gate(
    const float* __restrict__ vf, const float* __restrict__ keys,
    const float* __restrict__ rw, float* __restrict__ gates,
    int* __restrict__ cnt, int* __restrict__ tok, int* __restrict__ dst) {
  int t = blockIdx.x, v = blockIdx.y;
  const float* x = vf + ((size_t)v * T_ + t) * D_;
  __shared__ float lg[NEXP_];
  int lane = threadIdx.x;
  if (lane < NEXP_) {
    const float* ke = keys + (size_t)lane * D_;
    const float* re = rw + ((size_t)v * NEXP_ + lane) * D_;
    float a = 0.f, b = 0.f, c = 0.f;
    for (int d = 0; d < D_; ++d) {
      a += x[d] * ke[d]; b += ke[d] * ke[d]; c += x[d] * re[d];
    }
    lg[lane] = 2.f * a - b + c;
  }
  __syncthreads();
  if (lane == 0) {
    int   idx[TOPK_];
    float val[TOPK_];
    unsigned used = 0;
    for (int k = 0; k < TOPK_; ++k) {
      float best = -1e30f; int bi = 0;
      for (int e = 0; e < NEXP_; ++e)
        if (!((used >> e) & 1) && lg[e] > best) { best = lg[e]; bi = e; }
      used |= 1u << bi; idx[k] = bi; val[k] = best;
    }
    float mx = val[0], se = 0.f;
    for (int k = 0; k < TOPK_; ++k) { val[k] = __expf(val[k] - mx); se += val[k]; }
    for (int k = 0; k < TOPK_; ++k) {
      int e = idx[k];
      int le = v * NEXP_ + e;
      int slot = atomicAdd(&cnt[le], 1);
      int drow = ((v * T_ + t) * TOPK_) + k;
      tok[le * T_ + slot] = t;
      dst[le * T_ + slot] = drow;
      gates[drow] = val[k] / se;
    }
  }
}

// ------------------------------------------------- generic bf16 WMMA GEMM
// C[M,N] = act(A[M,K] * Bw[N,K]^T + bias[N]); optional row gather/scatter.
// One wave per 32x64 output tile: 8 accumulators, each B fragment feeds two
// WMMAs from registers -> 12 b128 loads per 8 v_wmma_f32_16x16x32_bf16.
__global__ __launch_bounds__(32) void k_gemm(
    const unsigned short* __restrict__ A, int lda,
    const unsigned short* __restrict__ Bw, int ldb,
    const float* __restrict__ bias,
    float* __restrict__ Cf, unsigned short* __restrict__ Cb, int ldc,
    const int* __restrict__ rows_in, const int* __restrict__ rows_out,
    const int* __restrict__ count_ptr,
    int M, int N, int K, int act) {
  int Meff = count_ptr ? *count_ptr : M;       // uniform scalar load
  int mt = blockIdx.y;
  if (mt * 32 >= Meff) return;                 // uniform early-out
  int nt0 = blockIdx.x * 64;

  int lane = threadIdx.x & 31;
  int half = lane >> 4;        // 0: lanes 0-15, 1: lanes 16-31
  int r    = lane & 15;

  // two 16-row A sub-tiles per wave
  const unsigned short* ap[2];
#pragma unroll
  for (int i = 0; i < 2; ++i) {
    int arow_i = mt * 32 + i * 16 + r;
    int arow = arow_i < Meff ? arow_i : (Meff - 1);     // clamp pad rows
    if (rows_in) arow = rows_in[arow];                  // gathered row
    ap[i] = A + (size_t)arow * lda + half * 8;
  }

  v8f acc[2][4] = {{{}, {}, {}, {}}, {{}, {}, {}, {}}};
  for (int k0 = 0; k0 < K; k0 += 32) {
    BF16x16 va[2];
#pragma unroll
    for (int i = 0; i < 2; ++i) {
      // A frag: lane r -> K[0..7],[16..23]; lane r+16 -> K[8..15],[24..31]
      uint4 c0 = *(const uint4*)(const void*)(ap[i] + k0);
      uint4 c1 = *(const uint4*)(const void*)(ap[i] + k0 + 16);
      va[i].u[0] = c0.x; va[i].u[1] = c0.y; va[i].u[2] = c0.z; va[i].u[3] = c0.w;
      va[i].u[4] = c1.x; va[i].u[5] = c1.y; va[i].u[6] = c1.z; va[i].u[7] = c1.w;
    }
#pragma unroll
    for (int j = 0; j < 4; ++j) {
      // B frag: lane n -> col n, K[0..7],[16..23]; lane n+16 -> col n, K[8..15],[24..31]
      const unsigned short* bp =
          Bw + (size_t)(nt0 + j * 16 + r) * ldb + k0 + half * 8;
      BF16x16 vb;
      uint4 d0 = *(const uint4*)(const void*)(bp);
      uint4 d1 = *(const uint4*)(const void*)(bp + 16);
      vb.u[0] = d0.x; vb.u[1] = d0.y; vb.u[2] = d0.z; vb.u[3] = d0.w;
      vb.u[4] = d1.x; vb.u[5] = d1.y; vb.u[6] = d1.z; vb.u[7] = d1.w;
      acc[0][j] = __builtin_amdgcn_wmma_f32_16x16x32_bf16(
          false, va[0].v, false, vb.v, (short)0, acc[0][j], false, false);
      acc[1][j] = __builtin_amdgcn_wmma_f32_16x16x32_bf16(
          false, va[1].v, false, vb.v, (short)0, acc[1][j], false, false);
    }
  }

  // epilogue: VGPR e -> row (base + e + half*8), col = nt0 + j*16 + (lane&15)
#pragma unroll
  for (int i = 0; i < 2; ++i) {
#pragma unroll
    for (int j = 0; j < 4; ++j) {
      int col = nt0 + j * 16 + r;
#pragma unroll
      for (int e = 0; e < 8; ++e) {
        int row_i = mt * 32 + i * 16 + e + half * 8;
        if (row_i >= Meff) continue;
        int row = rows_out ? rows_out[row_i] : row_i;
        float x = acc[i][j][e];
        if (bias) x += bias[col];
        if (act == 1) x = fmaxf(x, 0.f);                                    // relu
        else if (act == 2) x = 0.5f * x * (1.f + erff(x * 0.7071067811865476f)); // gelu
        if (Cf) Cf[(size_t)row * ldc + col] = x;
        if (Cb) Cb[(size_t)row * ldc + col] = f2bf(x);
      }
    }
  }
}

// -------------------------- combine top-k expert outputs + positional enc
__global__ __launch_bounds__(256) void k_combine(
    const float* __restrict__ O, const float* __restrict__ gates,
    const float* __restrict__ frac, const float* __restrict__ pe_s,
    const float* __restrict__ ple_s, float* __restrict__ xf,
    unsigned short* __restrict__ xb) {
  int t = blockIdx.x;
  float fr = frac[t];
  float r1 = fmaxf(fr, 1.0f / RES_);
  int i1 = (int)roundf(r1 * RES_) - 1;
  float l2 = log2f(fr);
  float r2 = fminf(0.0025f * l2 * l2, 1.0f);
  r2 = fmaxf(r2, 1.0f / RES_);
  int i2 = (int)roundf(r2 * RES_) - 1;
  float ps = exp2f(pe_s[0]), pls = exp2f(ple_s[0]);
  for (int c = threadIdx.x; c < D_; c += 256) {
    float acc = 0.f;
#pragma unroll
    for (int v = 0; v < NVIEWS_; ++v)
#pragma unroll
      for (int k = 0; k < TOPK_; ++k) {
        int row = (v * T_ + t) * TOPK_ + k;
        acc += gates[row] * O[(size_t)row * D_ + c];
      }
    int cc = c < HALF_ ? c : c - HALF_;
    float iv = (float)(c < HALF_ ? i1 : i2);
    float ang = iv / powf(50.0f, 2.0f * (float)cc / (float)HALF_);
    float pv = (cc & 1) ? cosf(ang) : sinf(ang);
    acc += pv * (c < HALF_ ? ps : pls);
    xf[(size_t)t * D_ + c] = acc;
    xb[(size_t)t * D_ + c] = f2bf(acc);
  }
}

// ---------------------------------------------------- attention (L=8 tiny)
__global__ __launch_bounds__(256) void k_attn(
    const float* __restrict__ QKV, unsigned short* __restrict__ attnb) {
  int b = blockIdx.x, h = blockIdx.y;
  __shared__ float sP[L_][L_];
  int tid = threadIdx.x;
  if (tid < L_ * L_) {
    int qi = tid >> 3, ki = tid & 7;
    const float* q  = QKV + ((size_t)(b * L_ + qi) * 3 * D_) + h * DH_;
    const float* kk = QKV + ((size_t)(b * L_ + ki) * 3 * D_) + D_ + h * DH_;
    float s = 0.f;
    for (int d = 0; d < DH_; ++d) s += q[d] * kk[d];
    sP[qi][ki] = s * 0.08838834764831845f;   // 1/sqrt(128)
  }
  __syncthreads();
  if (tid < L_) {
    float mx = -1e30f;
    for (int j = 0; j < L_; ++j) mx = fmaxf(mx, sP[tid][j]);
    float e[L_], se = 0.f;
    for (int j = 0; j < L_; ++j) { e[j] = __expf(sP[tid][j] - mx); se += e[j]; }
    for (int j = 0; j < L_; ++j) sP[tid][j] = e[j] / se;
  }
  __syncthreads();
  for (int idx = tid; idx < L_ * DH_; idx += 256) {
    int qi = idx >> 7, d = idx & (DH_ - 1);
    float acc = 0.f;
#pragma unroll
    for (int l = 0; l < L_; ++l)
      acc += sP[qi][l] * QKV[((size_t)(b * L_ + l) * 3 * D_) + 2 * D_ + h * DH_ + d];
    attnb[(size_t)(b * L_ + qi) * D_ + h * DH_ + d] = f2bf(acc);
  }
}

// --------------------------------------------- residual add + layernorm
__global__ __launch_bounds__(256) void k_addln(
    const float* __restrict__ x, const float* __restrict__ dlt,
    const float* __restrict__ g, const float* __restrict__ bta,
    float* __restrict__ xo, unsigned short* __restrict__ xob) {
  int t = blockIdx.x, tid = threadIdx.x;
  __shared__ float red[256];
  __shared__ float stat;
  float v0 = x[(size_t)t * D_ + tid]       + dlt[(size_t)t * D_ + tid];
  float v1 = x[(size_t)t * D_ + tid + 256] + dlt[(size_t)t * D_ + tid + 256];
  red[tid] = v0 + v1;
  __syncthreads();
  for (int s = 128; s > 0; s >>= 1) {
    if (tid < s) red[tid] += red[tid + s];
    __syncthreads();
  }
  if (tid == 0) stat = red[0] * (1.0f / D_);
  __syncthreads();
  float mean = stat;
  float d0 = v0 - mean, d1 = v1 - mean;
  __syncthreads();
  red[tid] = d0 * d0 + d1 * d1;
  __syncthreads();
  for (int s = 128; s > 0; s >>= 1) {
    if (tid < s) red[tid] += red[tid + s];
    __syncthreads();
  }
  if (tid == 0) stat = rsqrtf(red[0] * (1.0f / D_) + 1e-5f);
  __syncthreads();
  float rstd = stat;
  float o0 = d0 * rstd * g[tid]       + bta[tid];
  float o1 = d1 * rstd * g[tid + 256] + bta[tid + 256];
  xo[(size_t)t * D_ + tid]        = o0;
  xo[(size_t)t * D_ + tid + 256]  = o1;
  xob[(size_t)t * D_ + tid]       = f2bf(o0);
  xob[(size_t)t * D_ + tid + 256] = f2bf(o1);
}

// ------------------------------------------------------- final x * frac
__global__ __launch_bounds__(256) void k_final(
    const float* __restrict__ x, const float* __restrict__ frac,
    float* __restrict__ out) {
  for (int i = blockIdx.x * blockDim.x + threadIdx.x; i < T_ * D_;
       i += gridDim.x * blockDim.x)
    out[i] = x[i] * frac[i >> 9];
}

// =========================================================== host driver
extern "C" void kernel_launch(void* const* d_in, const int* in_sizes, int n_in,
                              void* d_out, int out_size, void* d_ws, size_t ws_size,
                              hipStream_t stream) {
  (void)in_sizes; (void)n_in; (void)out_size; (void)ws_size;
  const int*   Z        = (const int*)d_in[0];
  const float* frac     = (const float*)d_in[1];
  const float* emb_m2v  = (const float*)d_in[2];
  const float* emb_mag  = (const float*)d_in[3];
  const float* emb_oly  = (const float*)d_in[4];
  const float* pw_m2v   = (const float*)d_in[5];
  const float* pb_m2v   = (const float*)d_in[6];
  const float* pw_mag   = (const float*)d_in[7];
  const float* pb_mag   = (const float*)d_in[8];
  const float* pw_oly   = (const float*)d_in[9];
  const float* pb_oly   = (const float*)d_in[10];
  const float* router_w = (const float*)d_in[11];
  const float* keys     = (const float*)d_in[12];
  const float* exp_w1   = (const float*)d_in[13];
  const float* exp_b1   = (const float*)d_in[14];
  const float* exp_w2   = (const float*)d_in[15];
  const float* exp_b2   = (const float*)d_in[16];
  const float* emb_sc   = (const float*)d_in[17];
  const float* pe_sc    = (const float*)d_in[18];
  const float* ple_sc   = (const float*)d_in[19];
  const float* qkv_w    = (const float*)d_in[20];
  const float* qkv_b    = (const float*)d_in[21];
  const float* out_w    = (const float*)d_in[22];
  const float* out_b    = (const float*)d_in[23];
  const float* ln1_w    = (const float*)d_in[24];
  const float* ln1_b    = (const float*)d_in[25];
  const float* ff1_w    = (const float*)d_in[26];
  const float* ff1_b    = (const float*)d_in[27];
  const float* ff2_w    = (const float*)d_in[28];
  const float* ff2_b    = (const float*)d_in[29];
  const float* ln2_w    = (const float*)d_in[30];
  const float* ln2_b    = (const float*)d_in[31];
  float* outp = (float*)d_out;

  unsigned char* ws = (unsigned char*)d_ws;
  size_t off = 0;
  auto take = [&](size_t bytes) -> void* {
    void* p = ws + off;
    off = (off + bytes + 255) & ~(size_t)255;
    return p;
  };
  unsigned short* w1b   = (unsigned short*)take((size_t)NEXP_ * DFF_ * D_ * 2);
  unsigned short* w2b   = (unsigned short*)take((size_t)NEXP_ * D_ * DFF_ * 2);
  unsigned short* qkvb  = (unsigned short*)take((size_t)NLAYERS_ * 3 * D_ * D_ * 2);
  unsigned short* outwb = (unsigned short*)take((size_t)NLAYERS_ * D_ * D_ * 2);
  unsigned short* ff1b  = (unsigned short*)take((size_t)NLAYERS_ * DFF_ * D_ * 2);
  unsigned short* ff2b  = (unsigned short*)take((size_t)NLAYERS_ * D_ * DFF_ * 2);
  float*          viewsf= (float*)take((size_t)NVIEWS_ * T_ * D_ * 4);
  unsigned short* viewsb= (unsigned short*)take((size_t)NVIEWS_ * T_ * D_ * 2);
  float*          gates = (float*)take((size_t)NVIEWS_ * T_ * TOPK_ * 4);
  int*            mcnt  = (int*)take((size_t)NVIEWS_ * NEXP_ * 4);
  int*            mtok  = (int*)take((size_t)NVIEWS_ * NEXP_ * T_ * 4);
  int*            mdst  = (int*)take((size_t)NVIEWS_ * NEXP_ * T_ * 4);
  unsigned short* Hb    = (unsigned short*)take((size_t)NVIEWS_ * T_ * TOPK_ * DFF_ * 2);
  float*          Of    = (float*)take((size_t)NVIEWS_ * T_ * TOPK_ * D_ * 4);
  float*          xf    = (float*)take((size_t)T_ * D_ * 4);
  unsigned short* xb    = (unsigned short*)take((size_t)T_ * D_ * 2);
  float*          QKV   = (float*)take((size_t)T_ * 3 * D_ * 4);
  unsigned short* attnb = (unsigned short*)take((size_t)T_ * D_ * 2);
  float*          proj  = (float*)take((size_t)T_ * D_ * 4);
  unsigned short* Hfb   = (unsigned short*)take((size_t)T_ * DFF_ * 2);
  float*          f2    = (float*)take((size_t)T_ * D_ * 4);

  auto cvt = [&](const float* s, unsigned short* d, int n) {
    k_f32_to_bf16<<<2048, 256, 0, stream>>>(s, d, n);
  };
  auto gemm = [&](const unsigned short* A, int lda, const unsigned short* Bw,
                  int ldb, const float* bias, float* Cf, unsigned short* Cb,
                  int ldc, const int* rin, const int* rout, const int* cnt,
                  int M, int N, int K, int act) {
    dim3 g(N / 64, (M + 31) / 32);
    k_gemm<<<g, 32, 0, stream>>>(A, lda, Bw, ldb, bias, Cf, Cb, ldc, rin, rout,
                                 cnt, M, N, K, act);
  };

  // 0) zero routing counters; convert big weights fp32 -> bf16 (once/call)
  k_zero_i32<<<1, 64, 0, stream>>>(mcnt, NVIEWS_ * NEXP_);
  cvt(exp_w1, w1b, NEXP_ * DFF_ * D_);
  cvt(exp_w2, w2b, NEXP_ * D_ * DFF_);
  cvt(qkv_w, qkvb, NLAYERS_ * 3 * D_ * D_);
  cvt(out_w, outwb, NLAYERS_ * D_ * D_);
  cvt(ff1_w, ff1b, NLAYERS_ * DFF_ * D_);
  cvt(ff2_w, ff2b, NLAYERS_ * D_ * DFF_);

  // 1) embed + project (3 views)
  k_embed<<<T_, 256, 0, stream>>>(Z, emb_m2v, 200, pw_m2v, pb_m2v, emb_sc,
                                  viewsf + 0 * (size_t)T_ * D_,
                                  viewsb + 0 * (size_t)T_ * D_);
  k_embed<<<T_, 256, 0, stream>>>(Z, emb_mag, 22, pw_mag, pb_mag, emb_sc,
                                  viewsf + 1 * (size_t)T_ * D_,
                                  viewsb + 1 * (size_t)T_ * D_);
  k_embed<<<T_, 256, 0, stream>>>(Z, emb_oly, 44, pw_oly, pb_oly, emb_sc,
                                  viewsf + 2 * (size_t)T_ * D_,
                                  viewsb + 2 * (size_t)T_ * D_);

  // 2) gating: top-4 experts per (view, token) + expert-grouped lists
  k_gate<<<dim3(T_, NVIEWS_), 32, 0, stream>>>(viewsf, keys, router_w, gates,
                                               mcnt, mtok, mdst);

  // 3) sparse MoE: gathered GEMMs only for selected experts (4x less FLOPs)
  for (int v = 0; v < NVIEWS_; ++v)
    for (int e = 0; e < NEXP_; ++e) {
      int le = v * NEXP_ + e;
      gemm(viewsb + (size_t)v * T_ * D_, D_, w1b + (size_t)e * DFF_ * D_, D_,
           exp_b1 + e * DFF_, nullptr, Hb, DFF_, mtok + le * T_, mdst + le * T_,
           mcnt + le, T_, DFF_, D_, /*gelu*/ 2);
    }
  for (int v = 0; v < NVIEWS_; ++v)
    for (int e = 0; e < NEXP_; ++e) {
      int le = v * NEXP_ + e;
      gemm(Hb, DFF_, w2b + (size_t)e * D_ * DFF_, DFF_, exp_b2 + e * D_, Of,
           nullptr, D_, mdst + le * T_, mdst + le * T_, mcnt + le, T_, D_, DFF_,
           0);
    }

  // 4) gate-weighted combine + positional encodings -> x
  k_combine<<<T_, 256, 0, stream>>>(Of, gates, frac, pe_sc, ple_sc, xf, xb);

  // 5) transformer encoder layers
  for (int n = 0; n < NLAYERS_; ++n) {
    gemm(xb, D_, qkvb + (size_t)n * 3 * D_ * D_, D_, qkv_b + n * 3 * D_, QKV,
         nullptr, 3 * D_, nullptr, nullptr, nullptr, T_, 3 * D_, D_, 0);
    k_attn<<<dim3(B_, HEADS_), 256, 0, stream>>>(QKV, attnb);
    gemm(attnb, D_, outwb + (size_t)n * D_ * D_, D_, out_b + n * D_, proj,
         nullptr, D_, nullptr, nullptr, nullptr, T_, D_, D_, 0);
    k_addln<<<T_, 256, 0, stream>>>(xf, proj, ln1_w + n * D_, ln1_b + n * D_,
                                    xf, xb);
    gemm(xb, D_, ff1b + (size_t)n * DFF_ * D_, D_, ff1_b + n * DFF_, nullptr,
         Hfb, DFF_, nullptr, nullptr, nullptr, T_, DFF_, D_, /*relu*/ 1);
    gemm(Hfb, DFF_, ff2b + (size_t)n * D_ * DFF_, DFF_, ff2_b + n * D_, f2,
         nullptr, D_, nullptr, nullptr, nullptr, T_, D_, DFF_, 0);
    k_addln<<<T_, 256, 0, stream>>>(xf, f2, ln2_w + n * D_, ln2_b + n * D_, xf,
                                    xb);
  }

  // 6) output = x * frac
  k_final<<<1024, 256, 0, stream>>>(xf, frac, outp);
}
